// TBAttention_41326175322452
// MI455X (gfx1250) — matrix operand order repaired and checked by reading, not compiled, not run on hardware
//
#include <hip/hip_runtime.h>

// ---------------------------------------------------------------------------
// TBAttention for MI455X (gfx1250, wave32, WMMA)
// B=4, S=2048, DIM=1024, H=8, HD=64, NUM_B=1024, INNER=512, TOPK=64
// ---------------------------------------------------------------------------

typedef __bf16 bf16_t;
typedef __attribute__((ext_vector_type(16))) __bf16 v16bf;
typedef __attribute__((ext_vector_type(8)))  float  v8f;

#define BDIM   1024
#define SEQ    2048
#define NBATCH 4
#define NHEAD  8
#define HD     64
#define NUMB   1024
#define INNER  512
#define TOPK   64
#define BH     (NBATCH*NHEAD)      // 32
#define MROWS  (NBATCH*SEQ)        // 8192

__device__ __forceinline__ v8f wmma_bf16(v16bf a, v16bf b, v8f c) {
  // v_wmma_f32_16x16x32_bf16  D = A(16x32) * B(32x16) + C(16x16 f32)
  return __builtin_amdgcn_wmma_f32_16x16x32_bf16(
      /*neg_a=*/false, a, /*neg_b=*/false, b,
      /*c_mod=*/(short)0, c, /*reuse_a=*/false, /*reuse_b=*/false);
}

// ---------------------------------------------------------------------------
// Pack a row-major f32 weight matrix (K x N), or transposed (N x K) source,
// into the per-lane bf16 B-fragment layout for v_wmma_f32_16x16x32_bf16.
// B-frag layout (32x16 bf16): lane = n_local + 16*kg, element e holds
// k_local = 8*kg + e (e<8), 16 + 8*kg + (e-8) (e>=8).  Dst entry: [(nt*KT + kt)*32 + lane][16]
// ---------------------------------------------------------------------------
__global__ __launch_bounds__(256)
void pack_b_kernel(const float* __restrict__ src, bf16_t* __restrict__ dst,
                   int K, int N, int transposed) {
  int tid  = blockIdx.x * blockDim.x + threadIdx.x;
  int lane = tid & 31;
  int tile = tid >> 5;
  int KT = K >> 5;           // K / 32
  int NT = N >> 4;           // N / 16
  if (tile >= KT * NT) return;
  int kt = tile % KT;
  int nt = tile / KT;
  int n  = nt * 16 + (lane & 15);
  int kg = lane >> 4;        // lane half: k runs at 8*kg and 16+8*kg
  bf16_t* out = dst + ((size_t)tile * 32 + lane) * 16;
#pragma unroll
  for (int e = 0; e < 16; ++e) {
    int k = kt * 32 + (e < 8 ? kg * 8 + e : 16 + kg * 8 + (e - 8));
    float v = transposed ? src[(size_t)n * K + k] : src[(size_t)k * N + n];
    out[e] = (bf16_t)v;
  }
}

// Extract diagonal of b_param: bdiag[n][d] = b_param[n][d][d]
__global__ __launch_bounds__(256)
void bdiag_kernel(const float* __restrict__ b_param, float* __restrict__ bdiag) {
  int tid = blockIdx.x * blockDim.x + threadIdx.x;   // 1024*64 threads
  int n = tid >> 6, d = tid & 63;
  bdiag[tid] = b_param[(size_t)n * HD * HD + (size_t)d * HD + d];
}

// ---------------------------------------------------------------------------
// GEMM1:  qv = x (8192 x 1024, f32) * w_qv_pack (bf16 frags)
// Wave register-blocking: each wave computes a 16x64 output tile
// (1 A-frag + 4 B-frags -> 4 WMMA per K-step of 32).
// A-frag (16x32 bf16): lane m = lane&15; per lane two contiguous runs of 8:
//   k = 8*(lane>>4) + 0..7  and  k = 16 + 8*(lane>>4) + 0..7
// Output split: cols [0,512) -> q bf16 head-layout, [512,1024) -> v f32.
// ---------------------------------------------------------------------------
__global__ __launch_bounds__(128)
void gemm_qv_kernel(const float* __restrict__ x, const bf16_t* __restrict__ wpack,
                    bf16_t* __restrict__ q, float* __restrict__ v) {
  const int lane = threadIdx.x & 31;
  const int wv   = threadIdx.x >> 5;
  const int NT4 = BDIM / 64, MT = MROWS / 16, KT = BDIM / 32;
  const int tile = blockIdx.x * 4 + wv;
  if (tile >= MT * NT4) return;
  const int mt = tile / NT4, ng = tile % NT4;
  const int kg = lane >> 4;
  const float* arow = x + (size_t)(mt * 16 + (lane & 15)) * BDIM;
  const v16bf* bp = (const v16bf*)wpack;

  v8f c[4] = {};
  for (int kt = 0; kt < KT; ++kt) {
    const float* a0 = arow + kt * 32 + 8 * kg;
    __builtin_prefetch(a0 + 128, 0, 0);         // global_prefetch_b8, ~4 tiles ahead
    v16bf a;
#pragma unroll
    for (int e = 0; e < 8; ++e) {
      a[e]     = (bf16_t)a0[e];
      a[8 + e] = (bf16_t)a0[16 + e];
    }
#pragma unroll
    for (int j = 0; j < 4; ++j) {
      v16bf b = bp[(size_t)((ng * 4 + j) * KT + kt) * 32 + lane];
      c[j] = wmma_bf16(a, b, c[j]);
    }
  }

  // C frag: lane n = lane&15, rows m = 8*(lane>>4) + vg
  const int t0 = mt * 16 + 8 * kg;
#pragma unroll
  for (int j = 0; j < 4; ++j) {
    const int n = (ng * 4 + j) * 16 + (lane & 15);
    if (n < INNER) {               // q
      int h = n >> 6, d = n & 63;
#pragma unroll
      for (int vg = 0; vg < 8; ++vg) {
        int t = t0 + vg, b = t >> 11, i = t & 2047;
        q[(((size_t)(b * NHEAD + h)) * SEQ + i) * HD + d] = (bf16_t)c[j][vg];
      }
    } else {                       // v
      int n2 = n - INNER;
      int h = n2 >> 6, d = n2 & 63;
#pragma unroll
      for (int vg = 0; vg < 8; ++vg) {
        int t = t0 + vg, b = t >> 11, i = t & 2047;
        v[(((size_t)(b * NHEAD + h)) * SEQ + i) * HD + d] = c[j][vg];
      }
    }
  }
}

// ---------------------------------------------------------------------------
// Fused: sim = q * k_paramT (WMMA) -> top-64 -> softmax -> bdiag gather -> *v
// One block = one (bh, 16-row tile). 8 waves: phase1 each wave does 8 N-tiles
// of sim; phase2/3 each wave owns 2 rows.
// ---------------------------------------------------------------------------
__global__ __launch_bounds__(256)
void brain_attn_kernel(const bf16_t* __restrict__ q, const float* __restrict__ vbuf,
                       const bf16_t* __restrict__ kpack, const float* __restrict__ bdiag,
                       bf16_t* __restrict__ out_inner) {
  __shared__ bf16_t simb[16][NUMB];   // 32 KB
  __shared__ float  tkv[16][TOPK];    //  4 KB
  __shared__ int    tki[16][TOPK];    //  4 KB

  const int lane = threadIdx.x & 31;
  const int wv   = threadIdx.x >> 5;
  const int bh   = blockIdx.x >> 7;           // 128 row-tiles per bh
  const int it   = blockIdx.x & 127;
  const int i0   = it * 16;
  const int kg   = lane >> 4;
  const float scale = 0.125f;                 // HD^-0.5

  // ---- phase 1: sim tile (16 x 1024) via WMMA ----
  const bf16_t* qrow = q + (((size_t)bh * SEQ) + i0 + (lane & 15)) * HD;
  v16bf a0, a1;
#pragma unroll
  for (int e = 0; e < 8; ++e) {               // kt = 0
    a0[e]     = qrow[8 * kg + e];
    a0[8 + e] = qrow[16 + 8 * kg + e];
  }
#pragma unroll
  for (int e = 0; e < 8; ++e) {               // kt = 1
    a1[e]     = qrow[32 + 8 * kg + e];
    a1[8 + e] = qrow[48 + 8 * kg + e];
  }
  for (int nn = 0; nn < 8; ++nn) {
    int nt = wv * 8 + nn;
    v8f c = {};
    v16bf b0 = ((const v16bf*)kpack)[(size_t)(nt * 2 + 0) * 32 + lane];
    v16bf b1 = ((const v16bf*)kpack)[(size_t)(nt * 2 + 1) * 32 + lane];
    c = wmma_bf16(a0, b0, c);
    c = wmma_bf16(a1, b1, c);
    int n  = nt * 16 + (lane & 15);
    int m0 = 8 * kg;
#pragma unroll
    for (int vg = 0; vg < 8; ++vg)
      simb[m0 + vg][n] = (bf16_t)(c[vg] * scale);
  }
  __syncthreads();

  // ---- phase 2+3: per-row top-64, softmax, gather-accumulate, * v ----
  for (int rr = 0; rr < 2; ++rr) {
    int r = wv * 2 + rr;
    // iterative max-extraction (wave32)
    for (int kk = 0; kk < TOPK; ++kk) {
      float best = -__builtin_inff();
      int   bidx = -1;
#pragma unroll 4
      for (int t = 0; t < 32; ++t) {
        int n = lane + 32 * t;
        float vv = (float)simb[r][n];
        if (vv > best) { best = vv; bidx = n; }
      }
#pragma unroll
      for (int off = 16; off > 0; off >>= 1) {
        float ov = __shfl_xor(best, off, 32);
        int   oi = __shfl_xor(bidx, off, 32);
        if (ov > best || (ov == best && oi < bidx)) { best = ov; bidx = oi; }
      }
      if (lane == 0) {
        tkv[r][kk] = best;
        tki[r][kk] = bidx;
        simb[r][bidx] = (bf16_t)(-__builtin_inff());   // remove winner
      }
      // same-wave LDS ops are in-order: next scan sees the update
    }
    // softmax over the 64 selected (descending, so max == tkv[r][0])
    float m  = tkv[r][0];
    float e0 = __expf(tkv[r][lane]      - m);
    float e1 = __expf(tkv[r][lane + 32] - m);
    float s  = e0 + e1;
#pragma unroll
    for (int off = 16; off > 0; off >>= 1) s += __shfl_xor(s, off, 32);
    float inv = 1.0f / s;
    tkv[r][lane]      = e0 * inv;
    tkv[r][lane + 32] = e1 * inv;

    // acc[d] = sum_k w_k * bdiag[idx_k][d], d handled as lane and lane+32
    float acc0 = 0.f, acc1 = 0.f;
    for (int k = 0; k < TOPK; ++k) {
      float wk = tkv[r][k];
      const float* bd = bdiag + (size_t)tki[r][k] * HD;
      acc0 += wk * bd[lane];
      acc1 += wk * bd[lane + 32];
    }
    int i = i0 + r;
    int b = bh >> 3, h = bh & 7;
    const float* vp = vbuf + (((size_t)bh * SEQ) + i) * HD;
    bf16_t* op = out_inner + (((size_t)(b * SEQ + i)) * NHEAD + h) * HD;
    op[lane]      = (bf16_t)(acc0 * vp[lane]);
    op[lane + 32] = (bf16_t)(acc1 * vp[lane + 32]);
  }
}

// ---------------------------------------------------------------------------
// GEMM3: out = out_inner (8192 x 512 bf16) * w_out_pack + b_out  -> f32
// Wave register-blocking: 16x64 output tile per wave, 4 WMMA per K-step.
// ---------------------------------------------------------------------------
__global__ __launch_bounds__(128)
void gemm_out_kernel(const bf16_t* __restrict__ oin, const bf16_t* __restrict__ wpack,
                     const float* __restrict__ b_out, float* __restrict__ out) {
  const int lane = threadIdx.x & 31;
  const int wv   = threadIdx.x >> 5;
  const int NT4 = BDIM / 64, MT = MROWS / 16, KT = INNER / 32;
  const int tile = blockIdx.x * 4 + wv;
  if (tile >= MT * NT4) return;
  const int mt = tile / NT4, ng = tile % NT4;
  const int kg = lane >> 4;
  const bf16_t* arow = oin + (size_t)(mt * 16 + (lane & 15)) * INNER;
  const v16bf* bp = (const v16bf*)wpack;

  v8f c[4] = {};
  for (int kt = 0; kt < KT; ++kt) {
    const bf16_t* a0 = arow + kt * 32 + 8 * kg;
    __builtin_prefetch(a0 + 128, 0, 0);
    v16bf a;
#pragma unroll
    for (int e = 0; e < 8; ++e) {
      a[e]     = a0[e];
      a[8 + e] = a0[16 + e];
    }
#pragma unroll
    for (int j = 0; j < 4; ++j) {
      v16bf b = bp[(size_t)((ng * 4 + j) * KT + kt) * 32 + lane];
      c[j] = wmma_bf16(a, b, c[j]);
    }
  }

  const int t0 = mt * 16 + 8 * kg;
#pragma unroll
  for (int j = 0; j < 4; ++j) {
    const int n  = (ng * 4 + j) * 16 + (lane & 15);
    const float bn = b_out[n];
#pragma unroll
    for (int vg = 0; vg < 8; ++vg) {
      int t = t0 + vg;
      out[(size_t)t * BDIM + n] = c[j][vg] + bn;
    }
  }
}

// ---------------------------------------------------------------------------
// Host launcher
// ---------------------------------------------------------------------------
extern "C" void kernel_launch(void* const* d_in, const int* in_sizes, int n_in,
                              void* d_out, int out_size, void* d_ws, size_t ws_size,
                              hipStream_t stream) {
  const float* x       = (const float*)d_in[0];
  const float* b_param = (const float*)d_in[1];
  const float* k_param = (const float*)d_in[2];
  const float* w_qv    = (const float*)d_in[3];
  const float* w_out   = (const float*)d_in[4];
  const float* b_out   = (const float*)d_in[5];
  // d_in[6] = top_k (== 64, compiled in)

  char*  ws  = (char*)d_ws;
  size_t off = 0;
  auto carve = [&](size_t bytes) -> void* {
    void* p = ws + off;
    off = (off + bytes + 255) & ~(size_t)255;
    return p;
  };
  bf16_t* wqv_pack  = (bf16_t*)carve((size_t)BDIM  * BDIM * sizeof(bf16_t)); // 2 MB
  bf16_t* wout_pack = (bf16_t*)carve((size_t)INNER * BDIM * sizeof(bf16_t)); // 1 MB
  bf16_t* kpack     = (bf16_t*)carve((size_t)NUMB  * HD   * sizeof(bf16_t)); // 128 KB
  float*  bdiag     = (float*) carve((size_t)NUMB  * HD   * sizeof(float));  // 256 KB
  bf16_t* qbuf      = (bf16_t*)carve((size_t)BH * SEQ * HD * sizeof(bf16_t)); // 8 MB
  float*  vbuf      = (float*) carve((size_t)BH * SEQ * HD * sizeof(float));  // 16 MB
  bf16_t* oin       = (bf16_t*)carve((size_t)MROWS * INNER * sizeof(bf16_t)); // 8 MB

  // --- packing / prep (once per launch, tiny vs GEMMs) ---
  pack_b_kernel<<<(64 * 32 * 32) / 256, 256, 0, stream>>>(w_qv,  wqv_pack, BDIM,  BDIM, 0);
  pack_b_kernel<<<(64 * 16 * 32) / 256, 256, 0, stream>>>(w_out, wout_pack, INNER, BDIM, 0);
  pack_b_kernel<<<(64 * 2  * 32) / 256, 256, 0, stream>>>(k_param, kpack, HD, NUMB, 1);
  bdiag_kernel<<<(NUMB * HD) / 256, 256, 0, stream>>>(b_param, bdiag);

  // --- GEMM1: qv projection (2048 blocks x 4 waves, 16x64 tile/wave) ---
  gemm_qv_kernel<<<(MROWS / 16) * (BDIM / 64) / 4, 128, 0, stream>>>(x, wqv_pack, qbuf, vbuf);

  // --- fused sim / top-k / softmax / gather / *v ---
  brain_attn_kernel<<<BH * (SEQ / 16), 256, 0, stream>>>(qbuf, vbuf, kpack, bdiag, oin);

  // --- GEMM3: output projection + bias ---
  gemm_out_kernel<<<(MROWS / 16) * (BDIM / 64) / 4, 128, 0, stream>>>(oin, wout_pack, b_out, (float*)d_out);
}
